// MaxViTBlock_5231270167257
// MI455X (gfx1250) — matrix operand order repaired and checked by reading, not compiled, not run on hardware
//
#include <hip/hip_runtime.h>
#include <math.h>

// ---------------- types for WMMA ----------------
typedef _Float16 h16 __attribute__((ext_vector_type(16)));
typedef float    f8  __attribute__((ext_vector_type(8)));

union Frag16 { h16 v; uint4 u[2]; };
union Pack4  { _Float16 h[4]; uint2 u; };

#define CCH   128
#define HH    224
#define WW    224
#define HWPIX (HH*WW)            // 50176
#define PIX   (2*HWPIX)          // 100352 rows
#define NWIN  2048               // 2 * 32 * 32 windows
#define NTOK  49

// CDNA5 async global->LDS copy (ASYNCcnt-tracked).  ISA: GLOBAL_LOAD_ASYNC_TO_LDS_B128.
// LDS address = low 32 bits of the generic pointer (wave-relative LDS offset).
#define USE_ASYNC 1
#if USE_ASYNC
#define ASYNC_B128(lds32, gaddr64) \
    asm volatile("global_load_async_to_lds_b128 %0, %1, off" :: "v"(lds32), "v"(gaddr64) : "memory")
#define ASYNC_WAIT() asm volatile("s_wait_asynccnt 0x0" ::: "memory")
#endif

__device__ __forceinline__ float gelu_exact(float x) {
    return 0.5f * x * (1.0f + erff(x * 0.70710678118654752f));
}

// ---------------- NCHW -> NHWC ----------------
__global__ void k_nchw2nhwc(const float* __restrict__ in, float* __restrict__ out) {
    int idx = blockIdx.x * 256 + threadIdx.x;            // per float4 along W
    int q = HWPIX / 4;
    if (idx >= 2 * CCH * q) return;
    int w4 = idx % q; int rest = idx / q;
    int c = rest % CCH; int b = rest / CCH;
    float4 v = reinterpret_cast<const float4*>(in)[(size_t)(b * CCH + c) * q + w4];
    float vals[4] = {v.x, v.y, v.z, v.w};
    int p0 = w4 * 4;
#pragma unroll
    for (int j = 0; j < 4; ++j)
        out[((size_t)b * HWPIX + p0 + j) * CCH + c] = vals[j];
}

// ---------------- weight prep: [K][N] fp32 -> [N][K] f16 ----------------
__global__ void k_prep_wt(const float* __restrict__ in, _Float16* __restrict__ out,
                          int N, int K) {
    int idx = blockIdx.x * 256 + threadIdx.x;
    if (idx >= N * K) return;
    int n = idx / K, k = idx % K;
    out[(size_t)n * K + k] = (_Float16)in[(size_t)k * N + n];
}

// ---------------- row LayerNorm (C=128), optional GELU ----------------
template <bool DO_GELU>
__global__ void k_layernorm(const float* __restrict__ in, const float* __restrict__ w,
                            const float* __restrict__ b, float* __restrict__ out,
                            int rows) {
    int lane = threadIdx.x & 31;
    int row = blockIdx.x * 8 + (threadIdx.x >> 5);
    if (row >= rows) return;
    float4 v = reinterpret_cast<const float4*>(in + (size_t)row * CCH)[lane];
    float s  = v.x + v.y + v.z + v.w;
    float ss = v.x * v.x + v.y * v.y + v.z * v.z + v.w * v.w;
#pragma unroll
    for (int m = 1; m < 32; m <<= 1) { s += __shfl_xor(s, m, 32); ss += __shfl_xor(ss, m, 32); }
    float mu  = s * (1.0f / 128.0f);
    float var = ss * (1.0f / 128.0f) - mu * mu;
    float inv = rsqrtf(var + 1e-5f);
    float4 wv = reinterpret_cast<const float4*>(w)[lane];
    float4 bv = reinterpret_cast<const float4*>(b)[lane];
    float4 o;
    o.x = (v.x - mu) * inv * wv.x + bv.x;
    o.y = (v.y - mu) * inv * wv.y + bv.y;
    o.z = (v.z - mu) * inv * wv.z + bv.z;
    o.w = (v.w - mu) * inv * wv.w + bv.w;
    if (DO_GELU) { o.x = gelu_exact(o.x); o.y = gelu_exact(o.y); o.z = gelu_exact(o.z); o.w = gelu_exact(o.w); }
    reinterpret_cast<float4*>(out + (size_t)row * CCH)[lane] = o;
}

// ---------------- WMMA GEMM: out[M,N] = A[M,K] @ Bt[N,K]^T (+bias)(+resid)(gelu) --------
// A fp32 row-major; Bt = pre-converted f16 weights, [N][K] row-major (K contiguous).
// M % 128 == 0, N % 64 == 0, K % 32 == 0.  256 threads = 8 waves, wave tile 32x32.
template <bool HAS_BIAS, bool HAS_RES, bool DO_GELU>
__global__ __launch_bounds__(256)
void k_gemm(const float* __restrict__ A, const _Float16* __restrict__ Bt,
            const float* __restrict__ bias, const float* __restrict__ resid,
            float* __restrict__ out, int M, int N, int K) {
    __shared__ alignas(16) _Float16 As[128][40];   // [m][k]
    __shared__ alignas(16) _Float16 Bst[64][40];   // [n][k]
    int tid = threadIdx.x;
    int lane = tid & 31, wave = tid >> 5;
    int wm = wave & 3, wn = wave >> 2;
    int m0 = blockIdx.y * 128, n0 = blockIdx.x * 64;
    f8 acc[2][2] = {};

    for (int k0 = 0; k0 < K; k0 += 32) {
        // B tile 64x32 f16: one async b128 DMA per thread (64 rows x 4 chunks of 8 halfs)
        {
            int br = tid >> 2, bc = (tid & 3) * 8;
            const _Float16* gsrc = Bt + (size_t)(n0 + br) * K + k0 + bc;
#if USE_ASYNC
            unsigned lds32 = (unsigned)(size_t)&Bst[br][bc];
            unsigned long long ga = (unsigned long long)(size_t)gsrc;
            ASYNC_B128(lds32, ga);
#else
            *reinterpret_cast<uint4*>(&Bst[br][bc]) = *reinterpret_cast<const uint4*>(gsrc);
#endif
        }
        // stage A tile 128x32 (1024 float4), fp32 -> f16
#pragma unroll
        for (int i = 0; i < 4; ++i) {
            int idx = i * 256 + tid;
            int r = idx >> 3, c4 = idx & 7;
            float4 v = *reinterpret_cast<const float4*>(A + (size_t)(m0 + r) * K + k0 + c4 * 4);
            Pack4 pk;
            pk.h[0] = (_Float16)v.x; pk.h[1] = (_Float16)v.y;
            pk.h[2] = (_Float16)v.z; pk.h[3] = (_Float16)v.w;
            *reinterpret_cast<uint2*>(&As[r][c4 * 4]) = pk.u;
        }
        // prefetch next A K-slab (global_prefetch_b8)
        if (k0 + 32 < K)
            __builtin_prefetch(A + (size_t)(m0 + (tid >> 1)) * K + (k0 + 32) + (tid & 1) * 16, 0, 0);
#if USE_ASYNC
        ASYNC_WAIT();
#endif
        __syncthreads();

        Frag16 af[2], bf[2];
#pragma unroll
        for (int mi = 0; mi < 2; ++mi) {
            int m = wm * 32 + mi * 16 + (lane & 15);
            int kb = (lane < 16) ? 0 : 8;
            af[mi].u[0] = *reinterpret_cast<const uint4*>(&As[m][kb]);
            af[mi].u[1] = *reinterpret_cast<const uint4*>(&As[m][kb + 16]);
        }
#pragma unroll
        for (int nj = 0; nj < 2; ++nj) {
            int n = wn * 32 + nj * 16 + (lane & 15);
            int kb = (lane < 16) ? 0 : 8;
            bf[nj].u[0] = *reinterpret_cast<const uint4*>(&Bst[n][kb]);
            bf[nj].u[1] = *reinterpret_cast<const uint4*>(&Bst[n][kb + 16]);
        }
#pragma unroll
        for (int mi = 0; mi < 2; ++mi)
#pragma unroll
            for (int nj = 0; nj < 2; ++nj)
                acc[mi][nj] = __builtin_amdgcn_wmma_f32_16x16x32_f16(
                    false, af[mi].v, false, bf[nj].v, (short)0, acc[mi][nj], false, false);
        __syncthreads();
    }

    // epilogue (fully specialized, branch-free)
#pragma unroll
    for (int mi = 0; mi < 2; ++mi)
#pragma unroll
        for (int nj = 0; nj < 2; ++nj) {
            int nn = n0 + wn * 32 + nj * 16 + (lane & 15);
            float bv = HAS_BIAS ? bias[nn] : 0.0f;
#pragma unroll
            for (int r = 0; r < 8; ++r) {
                int mm = m0 + wm * 32 + mi * 16 + r + ((lane & 16) ? 8 : 0);
                float val = acc[mi][nj][r] + bv;
                if (HAS_RES) val += resid[(size_t)mm * N + nn];
                if (DO_GELU) val = gelu_exact(val);
                out[(size_t)mm * N + nn] = val;
            }
        }
}

// ---------------- depthwise 3x3, SAME, NHWC ----------------
__global__ void k_dwconv(const float* __restrict__ in, const float* __restrict__ wt,
                         const float* __restrict__ bias, float* __restrict__ out) {
    int idx = blockIdx.x * 256 + threadIdx.x;
    if (idx >= PIX * 32) return;
    int c4 = idx & 31; int pix = idx >> 5;
    int b = pix / HWPIX; int hw = pix % HWPIX;
    int h = hw / WW, w = hw % WW;
    int cb = c4 * 4;
    float a0 = 0, a1 = 0, a2 = 0, a3 = 0;
#pragma unroll
    for (int ky = 0; ky < 3; ++ky) {
        int hh = h + ky - 1;
        if (hh < 0 || hh >= HH) continue;
#pragma unroll
        for (int kx = 0; kx < 3; ++kx) {
            int ww = w + kx - 1;
            if (ww < 0 || ww >= WW) continue;
            float4 v = *reinterpret_cast<const float4*>(
                in + ((size_t)b * HWPIX + hh * WW + ww) * CCH + cb);
            a0 += v.x * wt[(cb + 0) * 9 + ky * 3 + kx];
            a1 += v.y * wt[(cb + 1) * 9 + ky * 3 + kx];
            a2 += v.z * wt[(cb + 2) * 9 + ky * 3 + kx];
            a3 += v.w * wt[(cb + 3) * 9 + ky * 3 + kx];
        }
    }
    float4 o;
    o.x = a0 + bias[cb + 0]; o.y = a1 + bias[cb + 1];
    o.z = a2 + bias[cb + 2]; o.w = a3 + bias[cb + 3];
    reinterpret_cast<float4*>(out)[idx] = o;
}

// ---------------- squeeze-excite ----------------
__global__ void k_se_partial(const float* __restrict__ y, float* __restrict__ part) {
    int c = threadIdx.x;                 // 128
    int chunk = blockIdx.x;              // 64
    int b = blockIdx.y;                  // 2
    size_t base = ((size_t)b * HWPIX + chunk * 784) * CCH + c;
    float s = 0.0f;
    for (int r = 0; r < 784; ++r) s += y[base + (size_t)r * CCH];
    part[(b * 64 + chunk) * CCH + c] = s;
}

__global__ void k_se_final(const float* __restrict__ part,
                           const float* __restrict__ fc1w, const float* __restrict__ fc1b,
                           const float* __restrict__ fc2w, const float* __restrict__ fc2b,
                           float* __restrict__ se) {
    __shared__ float mean[2][128];
    __shared__ float s1[2][32];
    int t = threadIdx.x; // 128
    for (int b = 0; b < 2; ++b) {
        float s = 0.0f;
        for (int ch = 0; ch < 64; ++ch) s += part[(b * 64 + ch) * CCH + t];
        mean[b][t] = s * (1.0f / (float)HWPIX);
    }
    __syncthreads();
    if (t < 64) {
        int b = t >> 5, o = t & 31;
        float s = fc1b[o];
        for (int c = 0; c < 128; ++c) s += mean[b][c] * fc1w[c * 32 + o];
        s1[b][o] = fmaxf(s, 0.0f);
    }
    __syncthreads();
    for (int b = 0; b < 2; ++b) {
        float s = fc2b[t];
        for (int o = 0; o < 32; ++o) s += s1[b][o] * fc2w[o * 128 + t];
        se[b * 128 + t] = 1.0f / (1.0f + __expf(-s));
    }
}

__global__ void k_scale(const float* __restrict__ in, const float* __restrict__ se,
                        float* __restrict__ out) {
    int idx = blockIdx.x * 256 + threadIdx.x;
    if (idx >= PIX * 32) return;
    int c4 = idx & 31; int b = (idx >> 5) / HWPIX;
    float4 v = reinterpret_cast<const float4*>(in)[idx];
    float4 s = reinterpret_cast<const float4*>(se)[b * 32 + c4];
    v.x *= s.x; v.y *= s.y; v.z *= s.z; v.w *= s.w;
    reinterpret_cast<float4*>(out)[idx] = v;
}

// ---------------- window/grid partition mapping ----------------
__device__ __forceinline__ size_t win_pixel(int row, int grid_mode) {
    int n = row % NTOK; int wc = row / NTOK;
    int gj = wc % 32; int t = wc / 32;
    int gi = t % 32; int b = t / 32;
    int wi = n / 7, wj = n % 7;
    int h, w;
    if (grid_mode) { h = wi * 32 + gi; w = wj * 32 + gj; }
    else           { h = gi * 7 + wi; w = gj * 7 + wj; }
    return (size_t)b * HWPIX + h * WW + w;
}

__global__ void k_winmap(const float* __restrict__ src, float* __restrict__ dst,
                         int grid_mode, int to_win) {
    int idx = blockIdx.x * 256 + threadIdx.x;
    if (idx >= PIX * 32) return;
    int c4 = idx & 31; int row = idx >> 5;
    size_t pix = win_pixel(row, grid_mode);
    if (to_win)
        reinterpret_cast<float4*>(dst)[(size_t)row * 32 + c4] =
            reinterpret_cast<const float4*>(src)[pix * 32 + c4];
    else
        reinterpret_cast<float4*>(dst)[pix * 32 + c4] =
            reinterpret_cast<const float4*>(src)[(size_t)row * 32 + c4];
}

__global__ void k_scatter_nchw(const float* __restrict__ src, float* __restrict__ out,
                               int grid_mode) {
    int idx = blockIdx.x * 256 + threadIdx.x;
    if (idx >= PIX * CCH) return;
    int c = idx & 127; int row = idx >> 7;
    size_t pix = win_pixel(row, grid_mode);
    int b = (int)(pix / HWPIX); int hw = (int)(pix % HWPIX);
    out[((size_t)(b * CCH + c)) * HWPIX + hw] = src[(size_t)row * CCH + c];
}

// ---------------- windowed attention (one block per window, one wave per head) ----------
__global__ __launch_bounds__(128)
void k_attn(const float* __restrict__ qkv, const float* __restrict__ rpb,
            float* __restrict__ outp) {
    // smemA: first used as q[h][64][32] (16KB) and k[h][64][32] (16KB),
    // later aliased as p[h][64][64] (32KB).  smemV: v^T [h][32][64] (16KB).
    __shared__ alignas(16) _Float16 smemA[2 * 4 * 64 * 32];
    __shared__ alignas(16) _Float16 smemV[4 * 32 * 64];
    _Float16* sq  = smemA;
    _Float16* sk  = smemA + 4 * 64 * 32;
    _Float16* sp  = smemA;           // aliases q|k after they are consumed
    _Float16* svt = smemV;

    int tid = threadIdx.x;
    int lane = tid & 31;
    int h = tid >> 5;                // head = wave
    int win = blockIdx.x;
    size_t r0 = (size_t)win * NTOK;

    // zero LDS (pad rows/cols must be 0)
    {
        unsigned int* za = reinterpret_cast<unsigned int*>(smemA);
        for (int i = tid; i < 4 * 64 * 32; i += 128) za[i] = 0u;   // 32KB as u32
        unsigned int* zv = reinterpret_cast<unsigned int*>(smemV);
        for (int i = tid; i < 2 * 32 * 64; i += 128) zv[i] = 0u;   // 16KB as u32
    }
    __syncthreads();

    // stage qkv rows (49 x 384 floats) -> f16 LDS; fold SCALE=0.5 into q
    for (int idx = tid; idx < NTOK * 96; idx += 128) {
        int n = idx / 96;
        int cq = (idx % 96) * 4;
        float4 v = *reinterpret_cast<const float4*>(qkv + (r0 + n) * 384 + cq);
        int sec = cq >> 7;            // 0:q 1:k 2:v
        int hh  = (cq >> 5) & 3;
        int d   = cq & 31;
        if (sec == 0) {
            Pack4 pk;
            pk.h[0] = (_Float16)(v.x * 0.5f); pk.h[1] = (_Float16)(v.y * 0.5f);
            pk.h[2] = (_Float16)(v.z * 0.5f); pk.h[3] = (_Float16)(v.w * 0.5f);
            *reinterpret_cast<uint2*>(&sq[((hh * 64) + n) * 32 + d]) = pk.u;
        } else if (sec == 1) {
            Pack4 pk;
            pk.h[0] = (_Float16)v.x; pk.h[1] = (_Float16)v.y;
            pk.h[2] = (_Float16)v.z; pk.h[3] = (_Float16)v.w;
            *reinterpret_cast<uint2*>(&sk[((hh * 64) + n) * 32 + d]) = pk.u;
        } else {
            svt[((hh * 32) + d + 0) * 64 + n] = (_Float16)v.x;
            svt[((hh * 32) + d + 1) * 64 + n] = (_Float16)v.y;
            svt[((hh * 32) + d + 2) * 64 + n] = (_Float16)v.z;
            svt[((hh * 32) + d + 3) * 64 + n] = (_Float16)v.w;
        }
    }
    __syncthreads();

    // load all Q (A-op) and K (B-op) fragments before q/k LDS is re-used for p
    Frag16 aq[4], bk[4];
#pragma unroll
    for (int mi = 0; mi < 4; ++mi) {
        int m = mi * 16 + (lane & 15);
        int kb = (lane < 16) ? 0 : 8;
        aq[mi].u[0] = *reinterpret_cast<const uint4*>(&sq[((h * 64) + m) * 32 + kb]);
        aq[mi].u[1] = *reinterpret_cast<const uint4*>(&sq[((h * 64) + m) * 32 + kb + 16]);
    }
#pragma unroll
    for (int nj = 0; nj < 4; ++nj) {
        int n = nj * 16 + (lane & 15);
        int kb = (lane < 16) ? 0 : 8;
        bk[nj].u[0] = *reinterpret_cast<const uint4*>(&sk[((h * 64) + n) * 32 + kb]);
        bk[nj].u[1] = *reinterpret_cast<const uint4*>(&sk[((h * 64) + n) * 32 + kb + 16]);
    }
    __syncthreads();   // all waves done reading sq/sk; safe to alias as sp

    // scores (one 16-row slab at a time) + bias + masked softmax -> p (f16, LDS)
#pragma unroll
    for (int mi = 0; mi < 4; ++mi) {
        f8 sc[4] = {};
#pragma unroll
        for (int nj = 0; nj < 4; ++nj)
            sc[nj] = __builtin_amdgcn_wmma_f32_16x16x32_f16(
                false, aq[mi].v, false, bk[nj].v, (short)0, sc[nj], false, false);
#pragma unroll
        for (int r = 0; r < 8; ++r) {
            int m = mi * 16 + r + ((lane & 16) ? 8 : 0);   // query index
            float vals[4];
#pragma unroll
            for (int nj = 0; nj < 4; ++nj) {
                int n = nj * 16 + (lane & 15);             // key index
                float v = sc[nj][r];
                if (n >= NTOK) v = -1e30f;
                else if (m < NTOK) {
                    int di = (m / 7) - (n / 7) + 6;
                    int dj = (m % 7) - (n % 7) + 6;
                    v += rpb[(di * 13 + dj) * 4 + h];
                }
                vals[nj] = v;
            }
            float mx = fmaxf(fmaxf(vals[0], vals[1]), fmaxf(vals[2], vals[3]));
#pragma unroll
            for (int s = 1; s < 16; s <<= 1) mx = fmaxf(mx, __shfl_xor(mx, s, 32));
            float sum = 0.0f;
#pragma unroll
            for (int nj = 0; nj < 4; ++nj) { vals[nj] = __expf(vals[nj] - mx); sum += vals[nj]; }
#pragma unroll
            for (int s = 1; s < 16; s <<= 1) sum += __shfl_xor(sum, s, 32);
            float inv = 1.0f / sum;
#pragma unroll
            for (int nj = 0; nj < 4; ++nj) {
                int n = nj * 16 + (lane & 15);
                sp[((h * 64) + m) * 64 + n] = (_Float16)(vals[nj] * inv);
            }
        }
    }
    __syncthreads();

    // O = P(64x64) @ V(64x32)
    f8 oacc[4][2] = {};
#pragma unroll
    for (int kc = 0; kc < 64; kc += 32) {
        Frag16 bv[2];
#pragma unroll
        for (int dj = 0; dj < 2; ++dj) {
            int d = dj * 16 + (lane & 15);
            int kb = kc + ((lane < 16) ? 0 : 8);
            bv[dj].u[0] = *reinterpret_cast<const uint4*>(&svt[((h * 32) + d) * 64 + kb]);
            bv[dj].u[1] = *reinterpret_cast<const uint4*>(&svt[((h * 32) + d) * 64 + kb + 16]);
        }
#pragma unroll
        for (int mi = 0; mi < 4; ++mi) {
            Frag16 ap;
            int m = mi * 16 + (lane & 15);
            int kb = kc + ((lane < 16) ? 0 : 8);
            ap.u[0] = *reinterpret_cast<const uint4*>(&sp[((h * 64) + m) * 64 + kb]);
            ap.u[1] = *reinterpret_cast<const uint4*>(&sp[((h * 64) + m) * 64 + kb + 16]);
#pragma unroll
            for (int dj = 0; dj < 2; ++dj)
                oacc[mi][dj] = __builtin_amdgcn_wmma_f32_16x16x32_f16(
                    false, ap.v, false, bv[dj].v, (short)0, oacc[mi][dj], false, false);
        }
    }

    // store O rows < 49, channel = h*32 + d
#pragma unroll
    for (int mi = 0; mi < 4; ++mi)
#pragma unroll
        for (int dj = 0; dj < 2; ++dj)
#pragma unroll
            for (int r = 0; r < 8; ++r) {
                int m = mi * 16 + r + ((lane & 16) ? 8 : 0);
                if (m < NTOK) {
                    int d = dj * 16 + (lane & 15);
                    outp[(r0 + m) * CCH + h * 32 + d] = oacc[mi][dj][r];
                }
            }
}

// ================= host launch =================
extern "C" void kernel_launch(void* const* d_in, const int* in_sizes, int n_in,
                              void* d_out, int out_size, void* d_ws, size_t ws_size,
                              hipStream_t stream) {
    (void)in_sizes; (void)n_in; (void)out_size; (void)ws_size;
    auto p = [&](int i) { return (const float*)d_in[i]; };
    const float* x = p(0);

    const size_t PC = (size_t)PIX * CCH;
    float* W0 = (float*)d_ws;
    float* W1 = W0 + PC;
    float* W2 = W1 + PC;
    float* W3 = W2 + PC;
    float* W4 = W3 + PC;                         // PIX*512
    float* SEP = W4 + (size_t)PIX * 512;         // 2*64*128
    float* SEV = SEP + 2 * 64 * 128;             // 2*128
    _Float16* WT = (_Float16*)(SEV + 256);       // converted weights (f16, [N][K])

    // pre-convert + transpose all GEMM weights -> f16 [N][K]
    struct WtDesc { int idx, K, N; };
    const WtDesc wd[12] = {
        {3, 128, 128},  {7, 128, 128},  {14, 128, 128}, {16, 128, 128},
        {20, 128, 384}, {23, 128, 128}, {27, 128, 512}, {29, 512, 128},
        {33, 128, 384}, {36, 128, 128}, {40, 128, 512}, {42, 512, 128}};
    _Float16* wptr[48];
    {
        size_t off = 0;
        for (int i = 0; i < 12; ++i) {
            _Float16* dst = WT + off;
            wptr[wd[i].idx] = dst;
            int tot = wd[i].N * wd[i].K;
            k_prep_wt<<<(tot + 255) / 256, 256, 0, stream>>>(p(wd[i].idx), dst, wd[i].N, wd[i].K);
            off += (size_t)tot;
        }
    }

    const int EB = (PIX * 32 + 255) / 256;       // elementwise-f4 blocks (12544)
    const int LB = PIX / 8;                      // layernorm blocks
    const dim3 G2(2, PIX / 128), G6(6, PIX / 128), G8(8, PIX / 128);

    // ---------- MBConv ----------
    k_nchw2nhwc<<<EB, 256, 0, stream>>>(x, W0);
    k_layernorm<false><<<LB, 256, 0, stream>>>(W0, p(1), p(2), W1, PIX);
    k_gemm<true, false, false><<<G2, 256, 0, stream>>>(W1, wptr[3], p(4), nullptr, W2, PIX, 128, 128);
    k_dwconv<<<EB, 256, 0, stream>>>(W2, p(5), p(6), W1);
    k_gemm<false, false, false><<<G2, 256, 0, stream>>>(W1, wptr[7], nullptr, nullptr, W2, PIX, 128, 128);
    k_layernorm<true><<<LB, 256, 0, stream>>>(W2, p(8), p(9), W1, PIX);   // LN2d + GELU
    k_se_partial<<<dim3(64, 2), 128, 0, stream>>>(W1, SEP);
    k_se_final<<<1, 128, 0, stream>>>(SEP, p(10), p(11), p(12), p(13), SEV);
    k_scale<<<EB, 256, 0, stream>>>(W1, SEV, W2);
    k_gemm<true, false, false><<<G2, 256, 0, stream>>>(W0, wptr[16], p(17), nullptr, W1, PIX, 128, 128); // skip
    k_gemm<true, true, false><<<G2, 256, 0, stream>>>(W2, wptr[14], p(15), W1, W3, PIX, 128, 128);       // proj+skip

    // ---------- transformers ----------
    auto run_tr = [&](int base, int grid_mode, int final_nchw) {
        k_winmap<<<EB, 256, 0, stream>>>(W3, W0, grid_mode, 1);                    // gather -> t
        k_layernorm<false><<<LB, 256, 0, stream>>>(W0, p(base + 0), p(base + 1), W1, PIX);
        k_gemm<true, false, false><<<G6, 256, 0, stream>>>(W1, wptr[base + 2], p(base + 3), nullptr,
                                                           W4, PIX, 384, 128);     // qkv
        k_attn<<<NWIN, 128, 0, stream>>>(W4, p(base + 4), W1);                     // attention
        k_gemm<true, true, false><<<G2, 256, 0, stream>>>(W1, wptr[base + 5], p(base + 6), W0,
                                                          W2, PIX, 128, 128);      // proj + resid
        k_layernorm<false><<<LB, 256, 0, stream>>>(W2, p(base + 7), p(base + 8), W1, PIX);
        k_gemm<true, false, true><<<G8, 256, 0, stream>>>(W1, wptr[base + 9], p(base + 10), nullptr,
                                                          W4, PIX, 512, 128);      // mlp1 + GELU
        k_gemm<true, true, false><<<G2, 256, 0, stream>>>(W4, wptr[base + 11], p(base + 12), W2,
                                                          W1, PIX, 128, 512);      // mlp2 + resid
        if (final_nchw)
            k_scatter_nchw<<<(PIX * CCH + 255) / 256, 256, 0, stream>>>(W1, (float*)d_out, grid_mode);
        else
            k_winmap<<<EB, 256, 0, stream>>>(W1, W3, grid_mode, 0);                // scatter -> NHWC
    };

    run_tr(18, /*grid=*/0, /*final=*/0);   // block (window) transformer
    run_tr(31, /*grid=*/1, /*final=*/1);   // grid transformer -> d_out (NCHW)
}
